// ConvLSTMModel_85418309583256
// MI455X (gfx1250) — compile-verified, matrix operand order checked by reading, not run on hardware
//
#include <hip/hip_runtime.h>
#include <hip/hip_bf16.h>

// ---------------------------------------------------------------------------
// CDNA5 (gfx1250) implementation. Heavy layers (cnn1..cnn4, 1x2 convs) are
// GEMM-mapped onto v_wmma_f32_16x16x32_bf16 with 2x2 register blocking:
// each wave computes a 32(oc) x 32(col) tile, 4 WMMA per K-chunk of 32.
// Activations live in zero-padded bf16 tensors so the LDS staging loop is
// branchless; weights are pre-swizzled into per-lane fragment order so the
// K-loop is pure {4x ds_load_b128, 4x global_load_b128, 4x wmma} + ptr bumps.
// wave32 throughout.
// ---------------------------------------------------------------------------

typedef __attribute__((ext_vector_type(16))) __bf16 v16bf;
typedef __attribute__((ext_vector_type(8)))  float  v8f;

union FragU { uint4 q[2]; v16bf v; __bf16 e[16]; };

__device__ __forceinline__ __bf16 f2bf(float f) {
    unsigned u = __builtin_bit_cast(unsigned, f);
    unsigned short h = (unsigned short)((u + 0x7FFFu + ((u >> 16) & 1u)) >> 16); // RNE
    return __builtin_bit_cast(__bf16, h);
}
__device__ __forceinline__ float bf2f(__bf16 b) {
    unsigned u = ((unsigned)__builtin_bit_cast(unsigned short, b)) << 16;
    return __builtin_bit_cast(float, u);
}
__device__ __forceinline__ float sigmoidf_(float x) { return 1.0f / (1.0f + expf(-x)); }

// ---------------------------------------------------------------------------
// Weight pre-swizzle: f32 [OC][K] -> bf16 fragment order [OT2p][NC][2][32][16].
// One thread per 16-element (32B) fragment; zero-padded, so the GEMM K-loop
// needs no guards.
// ---------------------------------------------------------------------------
__global__ void packA_kernel(const float* __restrict__ w, __bf16* __restrict__ dst,
                             int OC, int K, int NC, long long nFrag)
{
    long long fid = (long long)blockIdx.x * blockDim.x + threadIdx.x;
    if (fid >= nFrag) return;
    const int lane  = (int)(fid & 31);
    const int sub   = (int)((fid >> 5) & 1);
    const int chunk = (int)((fid >> 6) % NC);
    const int t2    = (int)((fid >> 6) / NC);
    const int n   = lane & 15;
    const int hi8 = (lane >> 4) << 3;
    const int oc  = t2 * 32 + sub * 16 + n;

    FragU f;
    #pragma unroll
    for (int g = 0; g < 2; ++g)
        #pragma unroll
        for (int j = 0; j < 8; ++j) {
            const int k = chunk * 32 + g * 16 + hi8 + j;
            float v = (oc < OC && k < K) ? w[(size_t)oc * K + k] : 0.0f;
            f.e[g * 8 + j] = f2bf(v);
        }
    uint4* d = reinterpret_cast<uint4*>(dst) + fid * 2;
    d[0] = f.q[0];
    d[1] = f.q[1];
}

// ---------------------------------------------------------------------------
// Stage 1: conv_con (4x4, stride 2, VALID) + relu + 2x2 maxpool -> scon[bt][961]
// ---------------------------------------------------------------------------
__global__ void conv_con_kernel(const float* __restrict__ replay,
                                const float* __restrict__ cw,
                                const float* __restrict__ cb,
                                float* __restrict__ scon)
{
    const int bt = blockIdx.x;
    const float* x = replay + (size_t)bt * 8 * 128 * 128;   // channel 0
    float w[16];
    #pragma unroll
    for (int i = 0; i < 16; ++i) w[i] = cw[i];
    const float b = cb[0];

    for (int t = threadIdx.x; t < 31 * 31; t += blockDim.x) {
        const int i = t / 31, j = t % 31;
        float mx = -1e30f;
        #pragma unroll
        for (int di = 0; di < 2; ++di)
            #pragma unroll
            for (int dj = 0; dj < 2; ++dj) {
                const int ci = 2 * i + di, cj = 2 * j + dj;
                float s = b;
                #pragma unroll
                for (int p = 0; p < 4; ++p)
                    #pragma unroll
                    for (int q = 0; q < 4; ++q)
                        s += x[(2 * ci + p) * 128 + (2 * cj + q)] * w[p * 4 + q];
                s = s > 0.0f ? s : 0.0f;
                mx = fmaxf(mx, s);
            }
        scon[(size_t)bt * 961 + t] = mx;
    }
}

// ---------------------------------------------------------------------------
// Stage 2: fc_con -> z row 0 (bf16, padded layout [1936][232] per image)
// ---------------------------------------------------------------------------
__global__ void fc_con_kernel(const float* __restrict__ scon,
                              const float* __restrict__ fw,
                              const float* __restrict__ fb,
                              __bf16* __restrict__ z)
{
    const int bt = blockIdx.x;
    const float* xr = scon + (size_t)bt * 961;
    for (int o = threadIdx.x; o < 200; o += blockDim.x) {
        const float* wr = fw + (size_t)o * 961;
        float s = fb[o];
        for (int i = 0; i < 961; ++i) s += wr[i] * xr[i];
        z[(size_t)bt * 1936 * 232 + o] = f2bf(s);
    }
}

// ---------------------------------------------------------------------------
// Stage 3: embedding rows 1..1927 of z (each row = emb_table[0] or [1]) -> bf16
// ---------------------------------------------------------------------------
__global__ void emb_kernel(const float* __restrict__ replay,
                           const float* __restrict__ emb,
                           __bf16* __restrict__ z)
{
    const int row = blockIdx.x + 1;      // 1..1927
    const int bt  = blockIdx.y;

    const int starts[5] = {1, 5, 7, 12, 14};
    const int scales[5] = {4, 2, 5, 2, 1914};
    const int chans[5]  = {1, 3, 5, 6, 7};

    int k = 4;
    if (row < 5) k = 0;
    else if (row < 7) k = 1;
    else if (row < 12) k = 2;
    else if (row < 14) k = 3;
    const int j = row - starts[k];

    const float vf = replay[((size_t)bt * 8 + chans[k]) * 128 * 128];
    int v = (int)vf;
    if (v > scales[k]) v = 0;
    const int idx = (j == v && v != 0) ? 1 : 0;

    const float* src = emb + (size_t)idx * 200;
    __bf16* dst = z + ((size_t)bt * 1936 + row) * 232;
    for (int c = threadIdx.x; c < 200; c += blockDim.x) dst[c] = f2bf(src[c]);
}

// ---------------------------------------------------------------------------
// Stage 4-7: 1x2 conv + bias + relu (+ optional width-2 maxpool) as WMMA GEMM.
//   in : bf16 padded [ICp][ldIn] per image (zero-padded -> branchless staging)
//   apk: bf16 weights pre-swizzled [OT2p][NC][2][32][16]
//   out: bf16 padded [.. ][ldOut] per image
// Per wave: 32(oc) x 32(conv col) tile, 4 accumulators.
// LDS holds one 32(col) x 32(kk) B slab in fragment-transposed order:
//   ldsBt[nn][kk] = in[c*16 + kk/2][x0 + nn + (kk&1)]
// ---------------------------------------------------------------------------
__global__ __launch_bounds__(128)
void conv_wmma2_kernel(const __bf16* __restrict__ in,
                       const __bf16* __restrict__ apk,
                       const float* __restrict__ bias,
                       __bf16* __restrict__ out,
                       int ICp, int ldIn, int OC, int Wout, int ldOut,
                       long long imgStrideOut, int NC, int doPool)
{
    const int bt   = blockIdx.z;
    const int x0   = blockIdx.x * 32;
    const int wave = threadIdx.x >> 5;
    const int lane = threadIdx.x & 31;
    const int ot   = blockIdx.y * 4 + wave;       // 32-row oc tile index

    __shared__ __align__(16) __bf16 ldsBt[32][40]; // [col nn][kk], stride 40

    const int n   = lane & 15;
    const int hi8 = (lane >> 4) << 3;
    const __bf16* inB = in + (size_t)bt * ICp * ldIn;
    const uint4* aBase = reinterpret_cast<const uint4*>(apk)
                         + (size_t)ot * NC * 128 + lane * 2;

    v8f acc00 = {}, acc01 = {}, acc10 = {}, acc11 = {};

    for (int c = 0; c < NC; ++c) {
        __syncthreads();
        // ---- branchless staging: 512 packed u32 stores by 128 threads ----
        for (int t2 = threadIdx.x; t2 < 512; t2 += 128) {
            const int nn = t2 >> 4;      // 0..31 column within 32-wide tile
            const int tt = t2 & 15;      // ic within chunk
            const __bf16* row = inB + (size_t)(c * 16 + tt) * ldIn + x0 + nn;
            const unsigned lo = (unsigned)__builtin_bit_cast(unsigned short, row[0]);
            const unsigned hi = (unsigned)__builtin_bit_cast(unsigned short, row[1]);
            *reinterpret_cast<unsigned*>(&ldsBt[nn][tt * 2]) = lo | (hi << 16);
        }
        __syncthreads();

        if (c + 1 < NC) {
            __builtin_prefetch(aBase + 128, 0, 1);
            __builtin_prefetch(inB + (size_t)(c + 1) * 16 * ldIn + x0, 0, 1);
        }

        FragU A0, A1, B0, B1;
        A0.q[0] = aBase[0];  A0.q[1] = aBase[1];    // sub 0 (oc rows 0..15)
        A1.q[0] = aBase[64]; A1.q[1] = aBase[65];   // sub 1 (oc rows 16..31)
        const uint4* bP0 = reinterpret_cast<const uint4*>(&ldsBt[n][0]);
        const uint4* bP1 = reinterpret_cast<const uint4*>(&ldsBt[n + 16][0]);
        B0.q[0] = bP0[(hi8 >> 3)]; B0.q[1] = bP0[2 + (hi8 >> 3)];
        B1.q[0] = bP1[(hi8 >> 3)]; B1.q[1] = bP1[2 + (hi8 >> 3)];

        acc00 = __builtin_amdgcn_wmma_f32_16x16x32_bf16(false, A0.v, false, B0.v, (short)0, acc00, false, false);
        acc01 = __builtin_amdgcn_wmma_f32_16x16x32_bf16(false, A0.v, false, B1.v, (short)0, acc01, false, false);
        acc10 = __builtin_amdgcn_wmma_f32_16x16x32_bf16(false, A1.v, false, B0.v, (short)0, acc10, false, false);
        acc11 = __builtin_amdgcn_wmma_f32_16x16x32_bf16(false, A1.v, false, B1.v, (short)0, acc11, false, false);

        aBase += 128;
    }

    // ---- epilogue: bias + relu (+ pool via lane-xor shuffle), write bf16 ----
    const int mbase = (lane >> 4) << 3;     // C/D layout: M = r (+8 for hi lanes)
    __bf16* outB = out + (size_t)bt * imgStrideOut;
    v8f accs[2][2] = {{acc00, acc01}, {acc10, acc11}};

    #pragma unroll
    for (int mt = 0; mt < 2; ++mt)
        #pragma unroll
        for (int nt = 0; nt < 2; ++nt)
            #pragma unroll
            for (int r = 0; r < 8; ++r) {
                const int oc  = ot * 32 + mt * 16 + r + mbase;
                const int col = x0 + nt * 16 + n;
                float v = accs[mt][nt][r] + ((oc < OC) ? bias[oc] : 0.0f);
                v = v > 0.0f ? v : 0.0f;
                if (doPool) {
                    const float other = __shfl_xor(v, 1, 32);
                    const float pooled = fmaxf(v, other);
                    const int p = col >> 1;
                    if (((lane & 1) == 0) && p < Wout && oc < OC)
                        outB[(size_t)oc * ldOut + p] = f2bf(pooled);
                } else {
                    if (col < Wout && oc < OC)
                        outB[(size_t)oc * ldOut + col] = f2bf(v);
                }
            }
}

// ---------------------------------------------------------------------------
// Stage 8: fc_cnn: a4(bf16)[bt][2300] @ W^T(100x2300) + b -> seq f32 [bt][100]
// ---------------------------------------------------------------------------
__global__ void fc_cnn_kernel(const __bf16* __restrict__ a4,
                              const float* __restrict__ fw,
                              const float* __restrict__ fb,
                              float* __restrict__ seq)
{
    const int bt = blockIdx.x;
    const __bf16* xr = a4 + (size_t)bt * 2300;
    for (int o = threadIdx.x; o < 100; o += blockDim.x) {
        const float* wr = fw + (size_t)o * 2300;
        float s = fb[o];
        for (int i = 0; i < 2300; ++i) s += wr[i] * bf2f(xr[i]);
        seq[(size_t)bt * 100 + o] = s;
    }
}

// ---------------------------------------------------------------------------
// Stage 9: 32-step LSTM (B=2, hidden=100) + fc_out -> d_out (2x2 f32)
// ---------------------------------------------------------------------------
__global__ void lstm_kernel(const float* __restrict__ seq,
                            const float* __restrict__ wih,
                            const float* __restrict__ whh,
                            const float* __restrict__ bih,
                            const float* __restrict__ bhh,
                            const float* __restrict__ wout,
                            const float* __restrict__ bout,
                            float* __restrict__ outp)
{
    __shared__ float h[200], c[200], g[800];
    const int t = threadIdx.x;
    for (int i = t; i < 200; i += blockDim.x) { h[i] = 0.0f; c[i] = 0.0f; }
    __syncthreads();

    for (int step = 0; step < 32; ++step) {
        for (int gi = t; gi < 800; gi += blockDim.x) {
            const int b = gi / 400, gg = gi % 400;
            const float* x  = seq + (size_t)(b * 32 + step) * 100;
            const float* wi = wih + (size_t)gg * 100;
            const float* wh = whh + (size_t)gg * 100;
            const float* hb = h + b * 100;
            float s = bih[gg] + bhh[gg];
            for (int j = 0; j < 100; ++j) s += wi[j] * x[j] + wh[j] * hb[j];
            g[gi] = s;
        }
        __syncthreads();
        for (int i = t; i < 200; i += blockDim.x) {
            const int b = i / 100, j = i % 100;
            const float* gb = g + b * 400;
            const float ig = sigmoidf_(gb[j]);
            const float fg = sigmoidf_(gb[100 + j]);
            const float cg = tanhf(gb[200 + j]);
            const float og = sigmoidf_(gb[300 + j]);
            const float cn = fg * c[i] + ig * cg;
            c[i] = cn;
            h[i] = og * tanhf(cn);
        }
        __syncthreads();
    }

    if (t < 4) {
        const int b = t >> 1, o = t & 1;
        float s = bout[o];
        for (int j = 0; j < 100; ++j) s += wout[o * 100 + j] * h[b * 100 + j];
        outp[b * 2 + o] = s;
    }
}

// ---------------------------------------------------------------------------
// Host-side orchestration
// ---------------------------------------------------------------------------
extern "C" void kernel_launch(void* const* d_in, const int* in_sizes, int n_in,
                              void* d_out, int out_size, void* d_ws, size_t ws_size,
                              hipStream_t stream)
{
    const float* replay = (const float*)d_in[0];
    const float* emb    = (const float*)d_in[1];
    const float* ccw    = (const float*)d_in[2];
    const float* ccb    = (const float*)d_in[3];
    const float* fcw    = (const float*)d_in[4];
    const float* fcb    = (const float*)d_in[5];
    const float* w1 = (const float*)d_in[6];  const float* b1 = (const float*)d_in[7];
    const float* w2 = (const float*)d_in[8];  const float* b2 = (const float*)d_in[9];
    const float* w3 = (const float*)d_in[10]; const float* b3 = (const float*)d_in[11];
    const float* w4 = (const float*)d_in[12]; const float* b4 = (const float*)d_in[13];
    const float* fcnw = (const float*)d_in[14]; const float* fcnb = (const float*)d_in[15];
    const float* wih = (const float*)d_in[16];  const float* whh = (const float*)d_in[17];
    const float* bih = (const float*)d_in[18];  const float* bhh = (const float*)d_in[19];
    const float* wo  = (const float*)d_in[20];  const float* bo  = (const float*)d_in[21];
    (void)in_sizes; (void)n_in; (void)out_size; (void)ws_size;

    const int BT = 64;  // B=2, T=32

    // workspace carve (256B aligned slices)
    char* wsb = (char*)d_ws;
    auto carve = [&](size_t bytes) -> char* {
        char* p = wsb; wsb += (bytes + 255) & ~(size_t)255; return p;
    };
    // padded bf16 activation tensors: [BT][ICp][ld]
    const size_t zBytes  = (size_t)BT * 1936 * 232 * 2;   // z:  IC 1928->1936, W 200 -> 232
    const size_t a1Bytes = (size_t)BT * 976  * 136 * 2;   // a1: 964->976,  99 -> 136
    const size_t a2Bytes = (size_t)BT * 496  * 72  * 2;   // a2: 482->496,  49 -> 72
    const size_t a3Bytes = (size_t)BT * 256  * 40  * 2;   // a3: 241->256,  24 -> 40
    float*  scon = (float*)carve((size_t)BT * 961 * 4);
    __bf16* z    = (__bf16*)carve(zBytes);
    __bf16* a1   = (__bf16*)carve(a1Bytes);
    __bf16* a2   = (__bf16*)carve(a2Bytes);
    __bf16* a3   = (__bf16*)carve(a3Bytes);
    __bf16* a4   = (__bf16*)carve((size_t)BT * 100 * 23 * 2);  // compact
    float*  seq  = (float*)carve((size_t)BT * 100 * 4);
    // pre-swizzled bf16 weights: [OT2p][NC][2][32][16]
    __bf16* w1b = (__bf16*)carve((size_t)32 * 121 * 1024 * 2);
    __bf16* w2b = (__bf16*)carve((size_t)16 * 61  * 1024 * 2);
    __bf16* w3b = (__bf16*)carve((size_t)8  * 31  * 1024 * 2);
    __bf16* w4b = (__bf16*)carve((size_t)4  * 16  * 1024 * 2);

    // zero the padded activation tensors (memset node is graph-capturable)
    hipMemsetAsync(z,  0, zBytes,  stream);
    hipMemsetAsync(a1, 0, a1Bytes, stream);
    hipMemsetAsync(a2, 0, a2Bytes, stream);
    hipMemsetAsync(a3, 0, a3Bytes, stream);

    auto packLaunch = [&](const float* w, __bf16* wb, int OC, int K, int NC, int OT2p) {
        long long nFrag = (long long)OT2p * NC * 64;
        int blocks = (int)((nFrag + 255) / 256);
        packA_kernel<<<blocks, 256, 0, stream>>>(w, wb, OC, K, NC, nFrag);
    };
    packLaunch(w1, w1b, 964, 3856, 121, 32);
    packLaunch(w2, w2b, 482, 1928, 61,  16);
    packLaunch(w3, w3b, 241,  964, 31,  8);
    packLaunch(w4, w4b, 100,  482, 16,  4);

    conv_con_kernel<<<BT, 256, 0, stream>>>(replay, ccw, ccb, scon);
    fc_con_kernel<<<BT, 256, 0, stream>>>(scon, fcw, fcb, z);
    emb_kernel<<<dim3(1927, BT), 256, 0, stream>>>(replay, emb, z);

    // GEMM convs: grid = (ceil(Wconv/32), OT2p/4, BT), 128 threads (4 waves)
    conv_wmma2_kernel<<<dim3(7, 8, BT), 128, 0, stream>>>(
        z,  w1b, b1, a1, 1936, 232, 964, 99, 136, (long long)976 * 136, 121, 1);
    conv_wmma2_kernel<<<dim3(4, 4, BT), 128, 0, stream>>>(
        a1, w2b, b2, a2, 976, 136, 482, 49, 72,  (long long)496 * 72,  61, 1);
    conv_wmma2_kernel<<<dim3(2, 2, BT), 128, 0, stream>>>(
        a2, w3b, b3, a3, 496, 72, 241, 24, 40,   (long long)256 * 40,  31, 1);
    conv_wmma2_kernel<<<dim3(1, 1, BT), 128, 0, stream>>>(
        a3, w4b, b4, a4, 256, 40, 100, 23, 23,   (long long)100 * 23,  16, 0);

    fc_cnn_kernel<<<BT, 256, 0, stream>>>(a4, fcnw, fcnb, seq);
    lstm_kernel<<<1, 256, 0, stream>>>(seq, wih, whh, bih, bhh, wo, bo, (float*)d_out);
}
